// RGAT_21225728377315
// MI455X (gfx1250) — compile-verified, matrix operand order checked by reading, not run on hardware
//
#include <hip/hip_runtime.h>
#include <hip/hip_bf16.h>
#include <math.h>

#define NN   50000
#define EE   800000
#define GG   128
#define H1   64
#define H2   64
#define RR   8
#define NB   30
#define NEG_SLOPE 0.2f

typedef __attribute__((ext_vector_type(16))) _Float16 v16h;
typedef __attribute__((ext_vector_type(8)))  _Float16 v8h;
typedef __attribute__((ext_vector_type(8)))  float    v8f;

// ---------- helpers ----------

__device__ __forceinline__ int f2ord(float f) {
    int i = __float_as_int(f);
    return (i >= 0) ? i : (i ^ 0x7fffffff);   // monotonic map for signed-int atomicMax
}
__device__ __forceinline__ float ord2f(int i) {
    return __int_as_float((i >= 0) ? i : (i ^ 0x7fffffff));
}

// Build a 16x32 f16 A-fragment (or 32x16 B-fragment, same per-lane K pattern)
// per CDNA5 wave32 layout: lanes 0-15 cover K={kb..kb+7, kb+16..kb+23} with kb=0,
// lanes 16-31 with kb=8.  rowbase points at element 0 of this lane's row/column.
__device__ __forceinline__ v16h load_frag(const _Float16* __restrict__ rowbase,
                                          int k0, int lane) {
    const int kb = (lane >> 4) << 3;                 // 0 or 8
    const _Float16* p = rowbase + k0 + kb;
    const v8h lo = *(const v8h*)(p);
    const v8h hi = *(const v8h*)(p + 16);
    v16h f;
#pragma unroll
    for (int i = 0; i < 8; ++i) { f[i] = lo[i]; f[i + 8] = hi[i]; }
    return f;
}

// ---------- prep kernels ----------

// Wt[r][h][g] (f16, col-major over g) = sum_b comp[r][b] * basis[b][g][h]
__global__ void k_prep_wt(const float* __restrict__ basis, const float* __restrict__ comp,
                          _Float16* __restrict__ Wt) {
    int idx = blockIdx.x * blockDim.x + threadIdx.x;
    if (idx >= RR * GG * H1) return;
    int r = idx / (GG * H1);
    int rem = idx - r * GG * H1;
    int g = rem / H1;
    int h = rem - g * H1;
    float acc = 0.f;
#pragma unroll 5
    for (int b = 0; b < NB; ++b)
        acc += comp[r * NB + b] * basis[(b * GG + g) * H1 + h];
    Wt[((size_t)r * H1 + h) * GG + g] = (_Float16)acc;
}

// Bt2[h2][k] (f16): k<64 -> W_rel[k][h2] ; k>=64 -> W_root[k-64][h2]
__global__ void k_prep_b2(const float* __restrict__ W_rel, const float* __restrict__ W_root,
                          _Float16* __restrict__ Bt2) {
    int idx = blockIdx.x * blockDim.x + threadIdx.x;
    if (idx >= H2 * 2 * H1) return;
    int h2 = idx >> 7;
    int k  = idx & 127;
    float v = (k < H1) ? W_rel[k * H2 + h2] : W_root[(k - H1) * H2 + h2];
    Bt2[h2 * 128 + k] = (_Float16)v;
}

__global__ void k_convert_x(const float* __restrict__ x, _Float16* __restrict__ xh) {
    int idx = blockIdx.x * blockDim.x + threadIdx.x;
    if (idx < NN * GG) xh[idx] = (_Float16)x[idx];
}

__global__ void k_init(float* __restrict__ x1, float* __restrict__ aggr,
                       float* __restrict__ denom, int* __restrict__ amax) {
    int idx = blockIdx.x * blockDim.x + threadIdx.x;
    if (idx < NN * H1) { x1[idx] = 0.f; aggr[idx] = 0.f; }
    if (idx < NN)      { denom[idx] = 0.f; amax[idx] = f2ord(-INFINITY); }
}

// ---------- GEMM 1: xw[r][n][h] = xh[n][:] . Wt[r][h][:]  (K = 128) ----------
// block = 128 threads = 4 waves; each wave computes one 16-row tile x 64 cols.
__global__ void k_gemm_xw(const _Float16* __restrict__ xh, const _Float16* __restrict__ Wt,
                          float* __restrict__ xw) {
    const int lane = threadIdx.x & 31;
    const int wave = threadIdx.x >> 5;
    const int r = blockIdx.y;
    const int mtile = blockIdx.x * 4 + wave;
    const int m0 = mtile * 16;
    if (m0 >= NN) return;                      // whole-wave uniform exit
    int mrow = m0 + (lane & 15);
    if (mrow >= NN) mrow = NN - 1;             // clamp loads; stores guarded
    const _Float16* Abase = xh + (size_t)mrow * GG;

    v8f acc[4] = {};
#pragma unroll
    for (int k0 = 0; k0 < GG; k0 += 32) {
        v16h a = load_frag(Abase, k0, lane);
#pragma unroll
        for (int c = 0; c < 4; ++c) {
            const int ncol = c * 16 + (lane & 15);
            const _Float16* Bbase = Wt + ((size_t)r * H1 + ncol) * GG;
            v16h b = load_frag(Bbase, k0, lane);
            acc[c] = __builtin_amdgcn_wmma_f32_16x16x32_f16(
                false, a, false, b, (short)0, acc[c], false, false);
        }
    }
    // C/D layout: lane n = lane%16 ; VGPR g -> row m0 + g + (lane>=16 ? 8 : 0)
    const int rowoff = (lane >> 4) * 8;
#pragma unroll
    for (int g = 0; g < 8; ++g) {
        int row = m0 + rowoff + g;
        if (row < NN) {
            float* out = xw + ((size_t)r * NN + row) * H1;
#pragma unroll
            for (int c = 0; c < 4; ++c) out[c * 16 + (lane & 15)] = acc[c][g];
        }
    }
}

// ---------- attention scalar dots: qdot[r][n], kdot[r][n] ----------
// one wave per (r,n); each lane handles h=lane and h=lane+32.
__global__ void k_dots(const float* __restrict__ xw, const float* __restrict__ q_att,
                       const float* __restrict__ k_att, float* __restrict__ qdot,
                       float* __restrict__ kdot) {
    const int lane = threadIdx.x & 31;
    const int w = (blockIdx.x * (blockDim.x >> 5)) + (threadIdx.x >> 5);
    if (w >= RR * NN) return;
    const float* row = xw + (size_t)w * H1;
    float v0 = row[lane], v1 = row[lane + 32];
    float qs = v0 * q_att[lane] + v1 * q_att[lane + 32];
    float ks = v0 * k_att[lane] + v1 * k_att[lane + 32];
#pragma unroll
    for (int off = 16; off >= 1; off >>= 1) {
        qs += __shfl_xor(qs, off, 32);
        ks += __shfl_xor(ks, off, 32);
    }
    if (lane == 0) { qdot[w] = qs; kdot[w] = ks; }
}

// ---------- edge pass 1: logits + segment max ----------
__global__ void k_alpha1(const int* __restrict__ ei, const int* __restrict__ et,
                         const float* __restrict__ qdot, const float* __restrict__ kdot,
                         float* __restrict__ alpha, int* __restrict__ amax) {
    int e = blockIdx.x * blockDim.x + threadIdx.x;
    if (e >= EE) return;
    int s = ei[e], d = ei[EE + e], t = et[e];
    float a = qdot[(size_t)t * NN + d] + kdot[(size_t)t * NN + s];
    a = (a > 0.f) ? a : NEG_SLOPE * a;
    alpha[e] = a;
    atomicMax(&amax[d], f2ord(a));
}

// ---------- edge pass 2: exp + denom ----------
__global__ void k_alpha2(const int* __restrict__ ei, const float* __restrict__ alpha,
                         const int* __restrict__ amax, float* __restrict__ ex,
                         float* __restrict__ denom) {
    int e = blockIdx.x * blockDim.x + threadIdx.x;
    if (e >= EE) return;
    int d = ei[EE + e];
    float m = ord2f(amax[d]);
    if (!isfinite(m)) m = 0.f;
    float v = __expf(alpha[e] - m);
    ex[e] = v;
    atomicAdd(&denom[d], v);
}

// ---------- edge pass 3: weighted scatter into x1 (64 lanes per edge) ----------
__global__ void k_scatter_x1(const int* __restrict__ ei, const int* __restrict__ et,
                             const float* __restrict__ xw, const float* __restrict__ ex,
                             const float* __restrict__ denom, float* __restrict__ x1) {
    int e = blockIdx.x * blockDim.y + threadIdx.y;
    if (e >= EE) return;
    int h = threadIdx.x;
    int s = ei[e], d = ei[EE + e], t = et[e];
    float coef = ex[e] / (denom[d] + 1e-16f);
    float v = xw[((size_t)t * NN + s) * H1 + h] * coef;
    atomicAdd(&x1[(size_t)d * H1 + h], v);
}

__global__ void k_bias(float* __restrict__ x1, const float* __restrict__ bias1) {
    int idx = blockIdx.x * blockDim.x + threadIdx.x;
    if (idx < NN * H1) x1[idx] += bias1[idx & (H1 - 1)];
}

// ---------- GraphConv neighbor aggregation ----------
__global__ void k_scatter_aggr(const int* __restrict__ ei, const float* __restrict__ x1,
                               float* __restrict__ aggr) {
    int e = blockIdx.x * blockDim.y + threadIdx.y;
    if (e >= EE) return;
    int h = threadIdx.x;
    int s = ei[e], d = ei[EE + e];
    atomicAdd(&aggr[(size_t)d * H1 + h], x1[(size_t)s * H1 + h]);
}

// Ah[n][k] f16: k<64 -> aggr[n][k] ; k>=64 -> x1[n][k-64]
__global__ void k_convert_A(const float* __restrict__ aggr, const float* __restrict__ x1,
                            _Float16* __restrict__ Ah) {
    int idx = blockIdx.x * blockDim.x + threadIdx.x;
    if (idx >= NN * 128) return;
    int n = idx >> 7, k = idx & 127;
    float v = (k < H1) ? aggr[(size_t)n * H1 + k] : x1[(size_t)n * H1 + (k - H1)];
    Ah[idx] = (_Float16)v;
}

// ---------- GEMM 2: out = [aggr|x1] @ [W_rel;W_root] + b_rel  (K = 128) ----------
__global__ void k_gemm_out(const _Float16* __restrict__ Ah, const _Float16* __restrict__ Bt2,
                           const float* __restrict__ b_rel, float* __restrict__ out) {
    const int lane = threadIdx.x & 31;
    const int wave = threadIdx.x >> 5;
    const int mtile = blockIdx.x * 4 + wave;
    const int m0 = mtile * 16;
    if (m0 >= NN) return;
    int mrow = m0 + (lane & 15);
    if (mrow >= NN) mrow = NN - 1;
    const _Float16* Abase = Ah + (size_t)mrow * 128;

    v8f acc[4] = {};
#pragma unroll
    for (int k0 = 0; k0 < 128; k0 += 32) {
        v16h a = load_frag(Abase, k0, lane);
#pragma unroll
        for (int c = 0; c < 4; ++c) {
            const int ncol = c * 16 + (lane & 15);
            const _Float16* Bbase = Bt2 + (size_t)ncol * 128;
            v16h b = load_frag(Bbase, k0, lane);
            acc[c] = __builtin_amdgcn_wmma_f32_16x16x32_f16(
                false, a, false, b, (short)0, acc[c], false, false);
        }
    }
    const int rowoff = (lane >> 4) * 8;
#pragma unroll
    for (int g = 0; g < 8; ++g) {
        int row = m0 + rowoff + g;
        if (row < NN) {
            float* o = out + (size_t)row * H2;
#pragma unroll
            for (int c = 0; c < 4; ++c) {
                int col = c * 16 + (lane & 15);
                o[col] = acc[c][g] + b_rel[col];
            }
        }
    }
}

// ---------- host ----------

extern "C" void kernel_launch(void* const* d_in, const int* in_sizes, int n_in,
                              void* d_out, int out_size, void* d_ws, size_t ws_size,
                              hipStream_t stream) {
    const float* node_features = (const float*)d_in[0];
    const int*   edge_index    = (const int*)d_in[1];   // [2,E]
    const int*   edge_type     = (const int*)d_in[2];   // [E]
    const float* basis         = (const float*)d_in[3];
    const float* comp          = (const float*)d_in[4];
    const float* q_att         = (const float*)d_in[5];
    const float* k_att         = (const float*)d_in[6];
    const float* bias1         = (const float*)d_in[7];
    const float* W_rel         = (const float*)d_in[8];
    const float* b_rel         = (const float*)d_in[9];
    const float* W_root        = (const float*)d_in[10];
    float* out = (float*)d_out;

    // carve workspace (256B aligned slices)
    char* ws = (char*)d_ws;
    size_t off = 0;
    auto carve = [&](size_t bytes) -> char* {
        char* p = ws + off;
        off = (off + bytes + 255) & ~(size_t)255;
        return p;
    };
    _Float16* xh    = (_Float16*)carve((size_t)NN * GG * 2);
    _Float16* Wt    = (_Float16*)carve((size_t)RR * H1 * GG * 2);
    _Float16* Bt2   = (_Float16*)carve((size_t)H2 * 128 * 2);
    float*    xw    = (float*)   carve((size_t)RR * NN * H1 * 4);
    float*    qdot  = (float*)   carve((size_t)RR * NN * 4);
    float*    kdot  = (float*)   carve((size_t)RR * NN * 4);
    int*      amax  = (int*)     carve((size_t)NN * 4);
    float*    denom = (float*)   carve((size_t)NN * 4);
    float*    alpha = (float*)   carve((size_t)EE * 4);
    float*    ex    = (float*)   carve((size_t)EE * 4);
    float*    x1    = (float*)   carve((size_t)NN * H1 * 4);
    float*    aggr  = (float*)   carve((size_t)NN * H1 * 4);
    _Float16* Ah    = (_Float16*)carve((size_t)NN * 128 * 2);
    (void)ws_size; (void)n_in; (void)in_sizes; (void)out_size;

    const int T = 256;
    // prep
    k_prep_wt <<<(RR * GG * H1 + T - 1) / T, T, 0, stream>>>(basis, comp, Wt);
    k_prep_b2 <<<(H2 * 128 + T - 1) / T, T, 0, stream>>>(W_rel, W_root, Bt2);
    k_convert_x<<<(NN * GG + T - 1) / T, T, 0, stream>>>(node_features, xh);
    k_init    <<<(NN * H1 + T - 1) / T, T, 0, stream>>>(x1, aggr, denom, amax);

    // xw GEMM: 3125 M-tiles, 4 per block, 8 relations
    dim3 g1((3125 + 3) / 4, RR);
    k_gemm_xw<<<g1, 128, 0, stream>>>(xh, Wt, xw);

    // attention scalars: R*N waves, 8 waves per block
    k_dots<<<(RR * NN + 7) / 8, 256, 0, stream>>>(xw, q_att, k_att, qdot, kdot);

    // edge softmax
    k_alpha1<<<(EE + T - 1) / T, T, 0, stream>>>(edge_index, edge_type, qdot, kdot, alpha, amax);
    k_alpha2<<<(EE + T - 1) / T, T, 0, stream>>>(edge_index, alpha, amax, ex, denom);

    // weighted message scatter
    dim3 bs(64, 4);
    k_scatter_x1<<<(EE + 3) / 4, bs, 0, stream>>>(edge_index, edge_type, xw, ex, denom, x1);
    k_bias<<<(NN * H1 + T - 1) / T, T, 0, stream>>>(x1, bias1);

    // GraphConv aggregation
    k_scatter_aggr<<<(EE + 3) / 4, bs, 0, stream>>>(edge_index, x1, aggr);

    // fused final GEMM
    k_convert_A<<<(NN * 128 + T - 1) / T, T, 0, stream>>>(aggr, x1, Ah);
    k_gemm_out<<<(3125 + 3) / 4, 128, 0, stream>>>(Ah, Bt2, b_rel, out);
}
